// AttentionLAI_91061896609800
// MI455X (gfx1250) — matrix operand order repaired and checked by reading, not compiled, not run on hardware
//
#include <hip/hip_runtime.h>

#define B_ 2
#define N_ 4096
#define C_ 384
#define H_ 6
#define D_ 64
#define SCALE_ 0.125f
#define NEG_ (-1e9f)

typedef __attribute__((ext_vector_type(16))) _Float16 v16h;
typedef __attribute__((ext_vector_type(8)))  float    v8f;

__device__ __forceinline__ v16h pun16(float4 a, float4 b) {
  union { float4 f[2]; v16h h; } u;
  u.f[0] = a; u.f[1] = b;
  return u.h;
}

__device__ __forceinline__ v16h cvt16(const float* v) {
  v16h r;
#pragma unroll
  for (int i = 0; i < 16; ++i) r[i] = (_Float16)v[i];
  return r;
}

__device__ __forceinline__ v8f wmma_f16(v16h a, v16h b, v8f c) {
  return __builtin_amdgcn_wmma_f32_16x16x32_f16(false, a, false, b, (short)0, c,
                                                false, false);
}

// ---------------------------------------------------------------------------
// Kernel 1: qkv = x @ qkv_w.T ; scatter into q[B,H,N,64], k[B,H,N,64],
// vT[B,H,64,N] as f16. One 16x64 output strip per wave (A tile reused by 4
// WMMAs). Strip-uniform values forced into SGPRs via readfirstlane.
// ---------------------------------------------------------------------------
__global__ __launch_bounds__(256) void k_qkv(const float* __restrict__ x,
                                             const float* __restrict__ w,
                                             _Float16* __restrict__ qh,
                                             _Float16* __restrict__ kh,
                                             _Float16* __restrict__ vT) {
  const int lane = threadIdx.x & 31;
  const int wv   = threadIdx.x >> 5;
  const int wid  = blockIdx.x * 8 + wv;
  const int NG   = (3 * C_) / 64;          // 18 column groups of 64
  const int m0 = __builtin_amdgcn_readfirstlane((wid / NG) * 16);
  const int j0 = __builtin_amdgcn_readfirstlane((wid % NG) * 64);
  const int n = lane & 15, g = lane >> 4;

  const float* arow = x + (size_t)(m0 + n) * C_;  // A: lane row m = lane&15
  const float* brow = w + (size_t)(j0 + n) * C_;  // B: lane col j = lane&15

  v8f acc[4] = {};
  for (int k0 = 0; k0 < C_; k0 += 32) {
    float4 ab[4];
    ab[0] = *(const float4*)(arow + k0 + 16 * g);
    ab[1] = *(const float4*)(arow + k0 + 16 * g + 4);
    ab[2] = *(const float4*)(arow + k0 + 16 * g + 8);
    ab[3] = *(const float4*)(arow + k0 + 16 * g + 12);
    const v16h A = cvt16((const float*)ab);
#pragma unroll
    for (int c = 0; c < 4; ++c) {
      const float* bc = brow + (size_t)(c * 16) * C_ + k0 + 16 * g;
      float4 bb[4];
      bb[0] = *(const float4*)(bc);
      bb[1] = *(const float4*)(bc + 4);
      bb[2] = *(const float4*)(bc + 8);
      bb[3] = *(const float4*)(bc + 12);
      acc[c] = wmma_f16(A, cvt16((const float*)bb), acc[c]);
    }
  }

  // j0 is a multiple of 64, so the whole 64-wide strip lies in one (t, head).
  const int t    = __builtin_amdgcn_readfirstlane(j0 / C_);
  const int head = __builtin_amdgcn_readfirstlane((j0 % C_) / D_);
#pragma unroll
  for (int c = 0; c < 4; ++c) {
    const int d = c * 16 + n;
#pragma unroll
    for (int r = 0; r < 8; ++r) {
      const int m  = m0 + r + 8 * g;
      const int b  = m >> 12;              // / N_
      const int nn = m & (N_ - 1);
      const size_t bh = (size_t)(b * H_ + head);
      const _Float16 val = (_Float16)acc[c][r];
      if (t == 0)      qh[(bh * N_ + nn) * D_ + d] = val;
      else if (t == 1) kh[(bh * N_ + nn) * D_ + d] = val;
      else             vT[(bh * D_ + d) * (size_t)N_ + nn] = val;
    }
  }
}

// ---------------------------------------------------------------------------
// Kernel 2: banded flash attention. One wave per (b, h, 16-query-row tile).
// Reads epoch on-device, restricts key chunks to the |i-j|<=w band.
// ---------------------------------------------------------------------------
__global__ __launch_bounds__(256) void k_attn(const _Float16* __restrict__ qh,
                                              const _Float16* __restrict__ kh,
                                              const _Float16* __restrict__ vT,
                                              _Float16* __restrict__ aout,
                                              const int* __restrict__ ep) {
  __shared__ _Float16 ldsP[8][16 * 32];    // per-wave P staging (f32->f16 transpose)
  const int lane = threadIdx.x & 31;
  const int wv   = threadIdx.x >> 5;
  const int wid  = blockIdx.x * 8 + wv;    // 0..3071
  const int q0   = __builtin_amdgcn_readfirstlane((wid & (N_ / 16 - 1)) * 16);
  const int bh   = __builtin_amdgcn_readfirstlane(wid >> 8);  // 0..11
  const int n = lane & 15, g = lane >> 4;

  // epoch schedule (device-side: epoch is a dynamic input)
  const int e = ep[0];
  const int w = (e >= 40) ? (1 << 29)
              : (e < 10) ? 8 : (e < 20) ? 12 : (e < 30) ? 18 : 20;
  int jlo = q0 - w;      if (jlo < 0)  jlo = 0;  jlo &= ~31;
  int jhi = q0 + 16 + w; if (jhi > N_) jhi = N_; jhi = (jhi + 31) & ~31;

  // Q as two 16x32 A-matrices (d = 0..31, 32..63), persistent in registers
  const _Float16* qrow = qh + ((size_t)bh * N_ + q0 + n) * D_;
  v16h Qa[2];
#pragma unroll
  for (int s = 0; s < 2; ++s)
    Qa[s] = pun16(*(const float4*)(qrow + 32 * s + 16 * g),
                  *(const float4*)(qrow + 32 * s + 16 * g + 8));

  float rmax[8], rsum[8];
  v8f O[4];
#pragma unroll
  for (int r = 0; r < 8; ++r) { rmax[r] = -3.0e38f; rsum[r] = 0.0f; }
#pragma unroll
  for (int dt = 0; dt < 4; ++dt) O[dt] = (v8f){};

  _Float16* pp = ldsP[wv];

  for (int jc = jlo; jc < jhi; jc += 32) {
    // prefetch next chunk's K/V cachelines (global_prefetch_b8)
    if (jc + 32 < jhi) {
      __builtin_prefetch(kh + ((size_t)bh * N_ + jc + 32 + lane) * D_, 0, 1);
      __builtin_prefetch(vT + ((size_t)bh * D_ + 2 * lane) * (size_t)N_ +
                             jc + 32, 0, 1);
    }
    // S = Q @ K^T for a 16x32 logit block (two 16x16 D tiles, 4 WMMAs)
    v8f S[2];
#pragma unroll
    for (int t = 0; t < 2; ++t) {
      const _Float16* krow = kh + ((size_t)bh * N_ + jc + 16 * t + n) * D_;
      v8f s = (v8f){};
#pragma unroll
      for (int ss = 0; ss < 2; ++ss) {
        v16h Bm = pun16(*(const float4*)(krow + 32 * ss + 16 * g),
                        *(const float4*)(krow + 32 * ss + 16 * g + 8));
        s = wmma_f16(Qa[ss], Bm, s);
      }
      S[t] = s;
    }
    // scale + band mask (f32, element-wise)
#pragma unroll
    for (int t = 0; t < 2; ++t)
#pragma unroll
      for (int r = 0; r < 8; ++r) {
        const int i  = q0 + r + 8 * g;
        const int j  = jc + 16 * t + n;
        const int dd = i - j;
        const bool in = (dd <= w) && (-dd <= w);
        S[t][r] = in ? S[t][r] * SCALE_ : NEG_;
      }
    // online softmax: row stats via shfl_xor across the 16-lane N-group
#pragma unroll
    for (int r = 0; r < 8; ++r) {
      float m2 = fmaxf(S[0][r], S[1][r]);
#pragma unroll
      for (int off = 8; off > 0; off >>= 1)
        m2 = fmaxf(m2, __shfl_xor(m2, off, 32));
      const float newm  = fmaxf(rmax[r], m2);
      const float alpha = __expf(rmax[r] - newm);
      const float p0 = __expf(S[0][r] - newm);
      const float p1 = __expf(S[1][r] - newm);
      float rs = p0 + p1;
#pragma unroll
      for (int off = 8; off > 0; off >>= 1)
        rs += __shfl_xor(rs, off, 32);
      rsum[r] = rsum[r] * alpha + rs;
      rmax[r] = newm;
#pragma unroll
      for (int dt = 0; dt < 4; ++dt) O[dt][r] *= alpha;
      // D-layout -> LDS row-major f16 P[16][32]
      pp[(r + 8 * g) * 32 + n]      = (_Float16)p0;
      pp[(r + 8 * g) * 32 + 16 + n] = (_Float16)p1;
    }
    __asm__ volatile("s_wait_dscnt 0" ::: "memory");  // wave-private staging, no barrier
    // reload P in 16x32 A-matrix layout
    const _Float16* prow = pp + n * 32;
    v16h Pa = pun16(*(const float4*)(prow + 16 * g),
                    *(const float4*)(prow + 16 * g + 8));
    // O += P @ V  (4 d-tiles; vT gives contiguous B operands)
#pragma unroll
    for (int dt = 0; dt < 4; ++dt) {
      const _Float16* vrow =
          vT + ((size_t)bh * D_ + dt * 16 + n) * (size_t)N_ + jc;
      v16h Vb = pun16(*(const float4*)(vrow + 16 * g),
                      *(const float4*)(vrow + 16 * g + 8));
      O[dt] = wmma_f16(Pa, Vb, O[dt]);
    }
  }

  // normalize and store to [B, N, C] f16
  const int b = bh / H_, head = bh % H_;
#pragma unroll
  for (int r = 0; r < 8; ++r) {
    const float inv = 1.0f / rsum[r];
    const size_t row = (size_t)b * N_ + q0 + r + 8 * g;
#pragma unroll
    for (int dt = 0; dt < 4; ++dt)
      aout[row * C_ + head * D_ + dt * 16 + n] = (_Float16)(O[dt][r] * inv);
  }
}

// ---------------------------------------------------------------------------
// Kernel 3: out = attn_out @ proj_w.T + proj_b  (f32 output), 16x64 strip/wave
// ---------------------------------------------------------------------------
__global__ __launch_bounds__(256) void k_proj(const _Float16* __restrict__ a,
                                              const float* __restrict__ w,
                                              const float* __restrict__ bias,
                                              float* __restrict__ out) {
  const int lane = threadIdx.x & 31;
  const int wv   = threadIdx.x >> 5;
  const int wid  = blockIdx.x * 8 + wv;
  const int NG   = C_ / 64;                // 6 column groups
  const int m0 = __builtin_amdgcn_readfirstlane((wid / NG) * 16);
  const int j0 = __builtin_amdgcn_readfirstlane((wid % NG) * 64);
  const int n = lane & 15, g = lane >> 4;

  const _Float16* arow = a + (size_t)(m0 + n) * C_;
  const float*    brow = w + (size_t)(j0 + n) * C_;

  v8f acc[4] = {};
  for (int k0 = 0; k0 < C_; k0 += 32) {
    const v16h A = pun16(*(const float4*)(arow + k0 + 16 * g),
                         *(const float4*)(arow + k0 + 16 * g + 8));
#pragma unroll
    for (int c = 0; c < 4; ++c) {
      const float* bc = brow + (size_t)(c * 16) * C_ + k0 + 16 * g;
      float4 bb[4];
      bb[0] = *(const float4*)(bc);
      bb[1] = *(const float4*)(bc + 4);
      bb[2] = *(const float4*)(bc + 8);
      bb[3] = *(const float4*)(bc + 12);
      acc[c] = wmma_f16(A, cvt16((const float*)bb), acc[c]);
    }
  }
#pragma unroll
  for (int c = 0; c < 4; ++c) {
    const float bv = bias[j0 + c * 16 + n];
#pragma unroll
    for (int r = 0; r < 8; ++r)
      out[(size_t)(m0 + r + 8 * g) * C_ + j0 + c * 16 + n] = acc[c][r] + bv;
  }
}

// ---------------------------------------------------------------------------
extern "C" void kernel_launch(void* const* d_in, const int* in_sizes, int n_in,
                              void* d_out, int out_size, void* d_ws,
                              size_t ws_size, hipStream_t stream) {
  (void)in_sizes; (void)n_in; (void)out_size; (void)ws_size;
  const float* x      = (const float*)d_in[0];
  const float* qkv_w  = (const float*)d_in[1];
  const float* proj_w = (const float*)d_in[2];
  const float* proj_b = (const float*)d_in[3];
  const int*   epoch  = (const int*)d_in[4];

  char* ws = (char*)d_ws;
  const size_t seg = (size_t)B_ * H_ * N_ * D_ * 2;  // 6 MB per f16 tensor
  _Float16* qh   = (_Float16*)(ws);
  _Float16* kh   = (_Float16*)(ws + seg);
  _Float16* vT   = (_Float16*)(ws + 2 * seg);
  _Float16* aout = (_Float16*)(ws + 3 * seg);        // [B,N,C] f16 (also 6 MB)
  float* out = (float*)d_out;

  // 512 row-tiles * 18 col-groups = 9216 waves / 8 waves-per-block
  k_qkv<<<1152, 256, 0, stream>>>(x, qkv_w, qh, kh, vT);
  // 12 (b*h) * 256 query-tiles = 3072 waves / 8
  k_attn<<<384, 256, 0, stream>>>(qh, kh, vT, aout, epoch);
  // 512 row-tiles * 6 col-groups = 3072 waves / 8
  k_proj<<<384, 256, 0, stream>>>(aout, proj_w, proj_b, out);
}
